// MoEAdapter_18777597018868
// MI455X (gfx1250) — matrix operand order repaired and checked by reading, not compiled, not run on hardware
//
#include <hip/hip_runtime.h>
#include <stdint.h>

// ---------------------------------------------------------------------------
// MoE top-2/8 FFN for MI455X (gfx1250, wave32, WMMA).
// Grouped-GEMM dispatch: gate -> per-expert token lists -> fused
// (X@W1 -> ReLU -> 263KB LDS bf16 hidden tile -> @W2 -> scale -> atomic add).
// MT=64 token tiles halve L2 weight traffic; 64x64 wave tiles give
// 16 v_wmma per 16 b128 loads in the steady state.
// ---------------------------------------------------------------------------

typedef __attribute__((ext_vector_type(16))) __bf16 bf16x16;
typedef __attribute__((ext_vector_type(8)))  float  f32x8;

constexpr int T    = 8 * 2048;   // tokens
constexpr int DIN  = 1024;
constexpr int DHID = 2048;
constexpr int DOUT = 1024;
constexpr int NEXP = 8;
constexpr int MT   = 64;         // tokens per expert tile
constexpr int HPAD = DHID + 8;   // LDS row stride: 1028 dwords == 4 mod 64
                                 // -> ds_read_b128 A-frags hit every bank 2x (minimum)

// workspace layout (bytes)
constexpr size_t O_COUNTS = 0;
constexpr size_t O_LIST   = 256;
constexpr size_t O_WLIST  = O_LIST  + sizeof(int)   * (size_t)NEXP * T;
constexpr size_t O_XB     = O_WLIST + sizeof(float) * (size_t)NEXP * T;
constexpr size_t O_W1B    = O_XB  + (size_t)T * DIN * 2;
constexpr size_t O_W2B    = O_W1B + (size_t)NEXP * DIN * DHID * 2;

#if __has_builtin(__builtin_amdgcn_global_load_async_to_lds_b32)
#define HAS_ASYNC_LDS 1
// builtin signature (from hipcc diagnostic): (AS1 int*, AS3 int*, imm, imm)
__device__ __forceinline__ void async_copy_b32(const void* g, void* l) {
  __builtin_amdgcn_global_load_async_to_lds_b32(
      (__attribute__((address_space(1))) int*)(g),
      (__attribute__((address_space(3))) int*)(l), 0, 0);
}
#endif

__device__ __forceinline__ unsigned short f2bf(float f) {
  unsigned int u = __builtin_bit_cast(unsigned int, f);
  u += 0x7FFFu + ((u >> 16) & 1u);   // round-to-nearest-even
  return (unsigned short)(u >> 16);
}

struct alignas(32) PairQ { uint4 lo; uint4 hi; };

// A-fragment, 16-bit 16x32 (ISA 7.12.2): lane&15 = row M; lane<16 holds
// K = kb+0..7 and kb+16..23, lane>=16 holds kb+8..15 and kb+24..31.
// caller passes p = row_ptr + kb + (lane>=16 ? 8 : 0)
__device__ __forceinline__ bf16x16 load_a_frag(const unsigned short* p) {
  PairQ f;
  f.lo = *(const uint4*)(p);
  f.hi = *(const uint4*)(p + 16);
  return __builtin_bit_cast(bf16x16, f);
}

// B-fragment, 16-bit 32x16: lane&15 = column N; lane<16 holds K=kb+0..7 and
// kb+16..23, lane>=16 holds kb+8..15 and kb+24..31 (matches A-fragment).
// caller passes p = col_ptr + kb + (lane>=16 ? 8 : 0)
__device__ __forceinline__ bf16x16 load_b_frag(const unsigned short* p) {
  PairQ f;
  f.lo = *(const uint4*)(p);
  f.hi = *(const uint4*)(p + 16);
  return __builtin_bit_cast(bf16x16, f);
}

// ---------------------------------------------------------------------------
__global__ void k_zero_counts(int* counts) {
  if (threadIdx.x < NEXP) counts[threadIdx.x] = 0;
}

// out = x (identity residual), Xb = bf16(x); fully coalesced float4 pass
__global__ void k_init(const float4* __restrict__ x4, float4* __restrict__ out4,
                       uint2* __restrict__ xb2, int n4) {
  int i = blockIdx.x * blockDim.x + threadIdx.x;
  if (i >= n4) return;
  float4 v = x4[i];
  out4[i] = v;
  uint2 p;
  p.x = (unsigned int)f2bf(v.x) | ((unsigned int)f2bf(v.y) << 16);
  p.y = (unsigned int)f2bf(v.z) | ((unsigned int)f2bf(v.w) << 16);
  xb2[i] = p;
}

// src fp32 [E][K][N] -> dst bf16 [E][N][K] (tiled LDS transpose + convert)
__global__ void k_transpose_bf16(const float* __restrict__ src,
                                 unsigned short* __restrict__ dst,
                                 int K, int N) {
  __shared__ float sm[32][33];
  int e = blockIdx.z;
  int n0 = blockIdx.x * 32, k0 = blockIdx.y * 32;
  const float* s = src + (size_t)e * K * N;
  unsigned short* d = dst + (size_t)e * N * K;
  int tx = threadIdx.x, ty = threadIdx.y;
#pragma unroll
  for (int j = 0; j < 32; j += 8)
    sm[ty + j][tx] = s[(size_t)(k0 + ty + j) * N + (n0 + tx)];
  __syncthreads();
#pragma unroll
  for (int j = 0; j < 32; j += 8)
    d[(size_t)(n0 + ty + j) * K + (k0 + tx)] = f2bf(sm[tx][ty + j]);
}

// gating: logits = x@gate_W, top-2 softmax renorm, scatter into expert lists
__global__ void k_gate(const float* __restrict__ x, const float* __restrict__ gw,
                       int* __restrict__ counts, int* __restrict__ list,
                       float* __restrict__ wlist) {
  int t = blockIdx.x * blockDim.x + threadIdx.x;
  if (t >= T) return;
  float acc[NEXP];
#pragma unroll
  for (int e = 0; e < NEXP; ++e) acc[e] = 0.0f;
  const float* xr = x + (size_t)t * DIN;
  for (int k = 0; k < DIN; ++k) {
    float xv = xr[k];
#pragma unroll
    for (int e = 0; e < NEXP; ++e) acc[e] += xv * gw[k * NEXP + e];
  }
  float m0 = -3.4e38f; int i0 = 0;
#pragma unroll
  for (int e = 0; e < NEXP; ++e)
    if (acc[e] > m0) { m0 = acc[e]; i0 = e; }
  float m1 = -3.4e38f; int i1 = 0;
#pragma unroll
  for (int e = 0; e < NEXP; ++e)
    if (e != i0 && acc[e] > m1) { m1 = acc[e]; i1 = e; }
  // renormalized top-2 softmax: g0 = e^{l0}/(e^{l0}+e^{l1})
  float g0 = 1.0f / (1.0f + __expf(m1 - m0));
  float g1 = 1.0f - g0;
  int s0 = atomicAdd(&counts[i0], 1);
  list[i0 * T + s0] = t; wlist[i0 * T + s0] = g0;
  int s1 = atomicAdd(&counts[i1], 1);
  list[i1 * T + s1] = t; wlist[i1 * T + s1] = g1;
}

// fused per-expert tile: GEMM1 -> ReLU -> LDS(bf16) -> GEMM2 -> scaled atomic
__launch_bounds__(256)
__global__ void k_moe(const unsigned short* __restrict__ Xb,
                      const unsigned short* __restrict__ W1b,
                      const float* __restrict__ b1,
                      const unsigned short* __restrict__ W2b,
                      const float* __restrict__ b2,
                      const int* __restrict__ counts,
                      const int* __restrict__ list,
                      const float* __restrict__ wlist,
                      float* __restrict__ out) {
  __shared__ __align__(16) unsigned short smH[MT][HPAD];  // 263KB hidden tile
  __shared__ int   smTok[MT];
  __shared__ float smW[MT];

  const int e = blockIdx.y;
  const int cnt = counts[e];
  const int tile = blockIdx.x;
  if (tile * MT >= cnt) return;   // uniform early exit (before any barrier)

  // ---- stage token ids + gate weights (CDNA5 async LDS DMA if available) ---
#if defined(HAS_ASYNC_LDS)
  if (threadIdx.x < MT) {  // waves 0,1: full EXEC
    int slot = tile * MT + threadIdx.x;
    int sslot = slot < cnt ? slot : (cnt - 1);   // clamp: valid global addr
    async_copy_b32(list + (size_t)e * T + sslot, &smTok[threadIdx.x]);
    async_copy_b32(wlist + (size_t)e * T + sslot, &smW[threadIdx.x]);
    asm volatile("s_wait_asynccnt 0x0" ::: "memory");
    if (slot >= cnt) smW[threadIdx.x] = 0.0f;    // tail rows contribute zero
  }
#else
  if (threadIdx.x < MT) {
    int slot = tile * MT + threadIdx.x;
    bool v = slot < cnt;
    smTok[threadIdx.x] = v ? list[(size_t)e * T + slot] : list[(size_t)e * T + cnt - 1];
    smW[threadIdx.x]   = v ? wlist[(size_t)e * T + slot] : 0.0f;
  }
#endif
  __syncthreads();

  const int wav = threadIdx.x >> 5, lane = threadIdx.x & 31;
  const int lh = lane & 15;
  const int aoff  = (lane >= 16) ? 8 : 0;
  const int boff  = (lane >= 16) ? 8 : 0;
  const int mhalf = (lane >= 16) ? 8 : 0;

  const unsigned short* xrow[4];
#pragma unroll
  for (int mi = 0; mi < 4; ++mi)
    xrow[mi] = Xb + (size_t)smTok[mi * 16 + lh] * DIN;

  const unsigned short* W1e = W1b + (size_t)e * DHID * DIN;   // [DHID][DIN]
  const unsigned short* W2e = W2b + (size_t)e * DOUT * DHID;  // [DOUT][DHID]

  // -------- GEMM1: H = relu(X @ W1 + b1); 8 waves x 64 cols, 4 passes ------
  for (int pass = 0; pass < DHID / 512; ++pass) {
    const int nb = pass * 512 + wav * 64;
    if (pass + 1 < DHID / 512)   // hint next weight slab toward the caches
      __builtin_prefetch(W1e + (size_t)(nb + 512) * DIN + lane * 64, 0, 0);
    f32x8 acc[4][4] = {};
    for (int kb = 0; kb < DIN; kb += 32) {
      bf16x16 af[4];
#pragma unroll
      for (int mi = 0; mi < 4; ++mi)
        af[mi] = load_a_frag(xrow[mi] + kb + aoff);
#pragma unroll
      for (int j = 0; j < 4; ++j) {
        bf16x16 bf = load_b_frag(W1e + (size_t)(nb + j * 16 + lh) * DIN + kb + boff);
#pragma unroll
        for (int mi = 0; mi < 4; ++mi)
          acc[mi][j] = __builtin_amdgcn_wmma_f32_16x16x32_bf16(
              false, af[mi], false, bf, (short)0, acc[mi][j], false, false);
      }
    }
#pragma unroll
    for (int j = 0; j < 4; ++j) {
      int n = nb + j * 16 + lh;
      float bias = b1[e * DHID + n];
#pragma unroll
      for (int mi = 0; mi < 4; ++mi)
#pragma unroll
        for (int i = 0; i < 8; ++i) {
          float v = acc[mi][j][i] + bias;
          v = v > 0.0f ? v : 0.0f;
          smH[mi * 16 + mhalf + i][n] = f2bf(v);
        }
    }
  }
  __syncthreads();

  // -------- GEMM2: out += w * (H @ W2 + b2); 2 passes over DOUT ------------
  for (int pass = 0; pass < DOUT / 512; ++pass) {
    const int nb = pass * 512 + wav * 64;
    if (pass + 1 < DOUT / 512)
      __builtin_prefetch(W2e + (size_t)(nb + 512) * DHID + lane * 64, 0, 0);
    f32x8 acc[4][4] = {};
    for (int kb = 0; kb < DHID; kb += 32) {
      bf16x16 af[4];
#pragma unroll
      for (int mi = 0; mi < 4; ++mi)
        af[mi] = load_a_frag(&smH[mi * 16 + lh][kb + aoff]);  // ds_read_b128
#pragma unroll
      for (int j = 0; j < 4; ++j) {
        bf16x16 bf = load_b_frag(W2e + (size_t)(nb + j * 16 + lh) * DHID + kb + boff);
#pragma unroll
        for (int mi = 0; mi < 4; ++mi)
          acc[mi][j] = __builtin_amdgcn_wmma_f32_16x16x32_bf16(
              false, af[mi], false, bf, (short)0, acc[mi][j], false, false);
      }
    }
#pragma unroll
    for (int j = 0; j < 4; ++j) {
      int n = nb + j * 16 + lh;
      float bias = b2[e * DOUT + n];
#pragma unroll
      for (int mi = 0; mi < 4; ++mi)
#pragma unroll
        for (int i = 0; i < 8; ++i) {
          int m = mi * 16 + mhalf + i;
          float val = (acc[mi][j][i] + bias) * smW[m];
          atomicAdd(&out[(size_t)smTok[m] * DOUT + n], val);
        }
    }
  }
}

// ---------------------------------------------------------------------------
extern "C" void kernel_launch(void* const* d_in, const int* in_sizes, int n_in,
                              void* d_out, int out_size, void* d_ws, size_t ws_size,
                              hipStream_t stream) {
  const float* x  = (const float*)d_in[0];
  const float* gw = (const float*)d_in[1];
  const float* W1 = (const float*)d_in[2];
  const float* b1 = (const float*)d_in[3];
  const float* W2 = (const float*)d_in[4];
  const float* b2 = (const float*)d_in[5];
  float* out = (float*)d_out;

  char* ws = (char*)d_ws;
  int*   counts = (int*)(ws + O_COUNTS);
  int*   list   = (int*)(ws + O_LIST);
  float* wlist  = (float*)(ws + O_WLIST);
  unsigned short* Xb  = (unsigned short*)(ws + O_XB);
  unsigned short* W1b = (unsigned short*)(ws + O_W1B);
  unsigned short* W2b = (unsigned short*)(ws + O_W2B);

  k_zero_counts<<<1, 32, 0, stream>>>(counts);

  // weights: fp32 [E][K][N] -> bf16 [E][N][K] (64MB total, L2-resident)
  k_transpose_bf16<<<dim3(DHID / 32, DIN / 32, NEXP), dim3(32, 8), 0, stream>>>(
      W1, W1b, DIN, DHID);
  k_transpose_bf16<<<dim3(DOUT / 32, DHID / 32, NEXP), dim3(32, 8), 0, stream>>>(
      W2, W2b, DHID, DOUT);

  int n4 = T * DIN / 4;
  k_init<<<(n4 + 255) / 256, 256, 0, stream>>>((const float4*)x, (float4*)out,
                                               (uint2*)Xb, n4);

  k_gate<<<(T + 255) / 256, 256, 0, stream>>>(x, gw, counts, list, wlist);

  // worst case: all tokens on one expert -> T/MT tiles; empty tiles exit early
  k_moe<<<dim3(T / MT, NEXP), 256, 0, stream>>>(Xb, W1b, b1, W2b, b2,
                                                counts, list, wlist, out);
}
